// MultiHeadedAttention_RPR_16355235463565
// MI455X (gfx1250) — matrix operand order repaired and checked by reading, not compiled, not run on hardware
//
#include <hip/hip_runtime.h>
#include <hip/hip_bf16.h>

// ---------------- types / helpers ----------------

typedef __attribute__((ext_vector_type(16))) __bf16 v16bf;
typedef __attribute__((ext_vector_type(8)))  float  v8f;

union Frag16 {
    v16bf v;
    unsigned int u[8];
};

__device__ __forceinline__ unsigned short f2bf(float x) {
    unsigned int u = __float_as_uint(x);
    unsigned int r = (u + 0x7FFFu + ((u >> 16) & 1u)) >> 16;   // RNE
    return (unsigned short)r;
}
__device__ __forceinline__ float bf2f(unsigned short s) {
    return __uint_as_float(((unsigned int)s) << 16);
}

// load 8 bf16 (16 bytes) into u[0..3] of a fragment half
__device__ __forceinline__ void load16B(unsigned int* dst, const unsigned short* src) {
    const uint4 t = *reinterpret_cast<const uint4*>(src);
    dst[0] = t.x; dst[1] = t.y; dst[2] = t.z; dst[3] = t.w;
}

// CDNA5 async direct global->LDS copy (no VGPR round-trip, tracked by ASYNCcnt).
// Generic LDS pointers carry the LDS byte offset in their low 32 bits
// (flat-aperture truncation rule), so we can feed the VDST operand directly.
__device__ __forceinline__ void async_copy16B(unsigned short* lds_dst,
                                              const unsigned short* gsrc) {
    unsigned int loff = (unsigned int)(unsigned long long)lds_dst;
    asm volatile("global_load_async_to_lds_b128 %0, %1, off"
                 :
                 : "v"(loff), "v"(gsrc)
                 : "memory");
}
__device__ __forceinline__ void wait_async0() {
    asm volatile("s_wait_asynccnt 0" ::: "memory");
}

__device__ __forceinline__ v8f wmma_bf16(const Frag16& a, const Frag16& b, v8f c) {
    return __builtin_amdgcn_wmma_f32_16x16x32_bf16(false, a.v, false, b.v,
                                                   (short)0, c, false, false);
}

__device__ __forceinline__ float redmax16(float v) {
    for (int off = 1; off < 16; off <<= 1)
        v = fmaxf(v, __shfl_xor(v, off, 32));
    return v;
}
__device__ __forceinline__ float redsum16(float v) {
    for (int off = 1; off < 16; off <<= 1)
        v += __shfl_xor(v, off, 32);
    return v;
}

#define S_LEN 2048
#define NB 2
#define NH 16
#define DK 64
#define DMODEL 1024
#define NBUCKET 33

// ---------------- fp32 -> bf16 convert ----------------

__global__ void cvt_bf16_kernel(const float* __restrict__ in,
                                unsigned short* __restrict__ out, int n) {
    int i = blockIdx.x * blockDim.x + threadIdx.x;
    if (i < n) out[i] = f2bf(in[i]);
}

// ---------------- generic WMMA GEMM:  Out = X(MxK) * W(NxK)^T + bias ----------------
// Double-buffered async global->LDS staging: prefetch of K-slab i+1 overlaps the
// 8 WMMAs of slab i; ASYNCcnt drained only after compute.
// mode 0: store bf16 into (B,H,S,dk) layout (projection kernels)
// mode 1: store fp32 row-major MxN (final output projection)

__global__ void __launch_bounds__(256)
gemm_bf16_kernel(const unsigned short* __restrict__ X,
                 const unsigned short* __restrict__ W,
                 const float* __restrict__ bias,
                 unsigned short* __restrict__ outb,
                 float* __restrict__ outf,
                 int M, int N, int K, int mode)
{
    __shared__ __align__(16) unsigned short Xs[2][128 * 32];
    __shared__ __align__(16) unsigned short Ws[2][128 * 32];

    const int tid  = threadIdx.x;
    const int lane = tid & 31;
    const int wave = tid >> 5;
    const int half = lane >> 4;
    const int l16  = lane & 15;
    const int wr   = wave >> 2;      // 0..1
    const int wc   = wave & 3;       // 0..3
    const int m0   = blockIdx.y * 128;
    const int n0   = blockIdx.x * 128;

    const int lrow = tid >> 1;            // 0..127
    const int lcol = (tid & 1) << 4;      // 0 or 16

    const unsigned short* xbase = X + (size_t)(m0 + lrow) * K + lcol;
    const unsigned short* wbase = W + (size_t)(n0 + lrow) * K + lcol;

    v8f zero = {};
    v8f acc[4][2];
    for (int i = 0; i < 4; ++i)
        for (int j = 0; j < 2; ++j) acc[i][j] = zero;

    auto stage = [&](int buf, int kk) {
        const unsigned short* xsrc = xbase + kk;
        unsigned short*       xdst = &Xs[buf][lrow * 32 + lcol];
        async_copy16B(xdst,     xsrc);
        async_copy16B(xdst + 8, xsrc + 8);
        const unsigned short* wsrc = wbase + kk;
        unsigned short*       wdst = &Ws[buf][lrow * 32 + lcol];
        async_copy16B(wdst,     wsrc);
        async_copy16B(wdst + 8, wsrc + 8);
    };

    stage(0, 0);
    wait_async0();
    __syncthreads();

    int cur = 0;
    for (int kk = 0; kk < K; kk += 32) {
        if (kk + 32 < K) stage(cur ^ 1, kk + 32);   // async prefetch of next slab

        const unsigned short* Xc = Xs[cur];
        const unsigned short* Wc = Ws[cur];
        Frag16 a4[4], b2[2];
        for (int i = 0; i < 4; ++i) {
            int m = wr * 64 + i * 16 + l16;
            // A frag: lane=row, elements {K base..base+7, base+16..base+23}, base = half*8
            load16B(a4[i].u + 0, Xc + m * 32 + half * 8);
            load16B(a4[i].u + 4, Xc + m * 32 + half * 8 + 16);
        }
        for (int j = 0; j < 2; ++j) {
            int n = wc * 32 + j * 16 + l16;
            // B frag: lane=col, elements {K base..base+7, base+16..base+23}, base = half*8
            load16B(b2[j].u + 0, Wc + n * 32 + half * 8);
            load16B(b2[j].u + 4, Wc + n * 32 + half * 8 + 16);
        }
        for (int i = 0; i < 4; ++i)
            for (int j = 0; j < 2; ++j)
                acc[i][j] = wmma_bf16(a4[i], b2[j], acc[i][j]);

        wait_async0();      // next slab landed (this wave)
        __syncthreads();    // all waves done reading cur + their prefetches landed
        cur ^= 1;
    }

    // epilogue
    for (int i = 0; i < 4; ++i) {
        for (int j = 0; j < 2; ++j) {
            for (int r = 0; r < 8; ++r) {
                int gm = m0 + wr * 64 + i * 16 + r + half * 8;
                int gn = n0 + wc * 32 + j * 16 + l16;
                float val = acc[i][j][r] + bias[gn];
                if (mode == 0) {
                    int s  = gm & (S_LEN - 1);
                    int bb = gm >> 11;          // / S_LEN
                    int h  = gn >> 6;           // / DK
                    int d  = gn & (DK - 1);
                    outb[(((size_t)bb * NH + h) * S_LEN + s) * DK + d] = f2bf(val);
                } else {
                    outf[(size_t)gm * N + gn] = val;
                }
            }
        }
    }
}

// ---------------- attention with relative position representations ----------------
// Grid: (S/128, B*H). Block: 256 threads = 8 waves; each wave owns 16 q-rows.

__global__ void __launch_bounds__(256)
attn_rpr_kernel(const unsigned short* __restrict__ Qp,   // (B,H,S,dk) bf16
                const unsigned short* __restrict__ Kp,
                const unsigned short* __restrict__ Vp,
                const float* __restrict__ embK,           // (33,dk)
                const float* __restrict__ embV,           // (33,dk)
                unsigned short* __restrict__ ctx)         // (B,S,H*dk) bf16
{
    __shared__ __align__(16) unsigned short Kt[64 * 64];        // [k][d]
    __shared__ __align__(16) unsigned short Vt[64 * 64];        // [d][k] (transposed)
    __shared__ __align__(16) unsigned short Pst[8][16 * 64];    // per-wave P staging
    __shared__ unsigned short relL[128 * NBUCKET];              // rel-K logits (bf16)
    __shared__ float wsum[128 * NBUCKET];                       // bucket-summed weights

    const int tid  = threadIdx.x;
    const int lane = tid & 31;
    const int wave = tid >> 5;
    const int half = lane >> 4;
    const int l16  = lane & 15;
    const int bh   = blockIdx.y;
    const int q0   = blockIdx.x * 128;

    const unsigned short* Qh = Qp + (size_t)bh * S_LEN * DK;
    const unsigned short* Kh = Kp + (size_t)bh * S_LEN * DK;
    const unsigned short* Vh = Vp + (size_t)bh * S_LEN * DK;

    for (int i = tid; i < 128 * NBUCKET; i += 256) wsum[i] = 0.0f;

    // relL[q_local][v] = Q[q0+q_local] . embK[v]
    for (int i = tid; i < 128 * NBUCKET; i += 256) {
        int ql = i / NBUCKET, v = i - ql * NBUCKET;
        const unsigned short* qr = Qh + (size_t)(q0 + ql) * DK;
        float a = 0.0f;
        for (int d = 0; d < DK; ++d) a += bf2f(qr[d]) * embK[v * DK + d];
        relL[i] = f2bf(a);
    }
    __syncthreads();

    // preload Q A-fragments (16 rows x 64 dk = 2 K-chunks of 32)
    Frag16 qA[2];
    {
        const unsigned short* qr = Qh + (size_t)(q0 + wave * 16 + l16) * DK;
        for (int kc = 0; kc < 2; ++kc) {
            load16B(qA[kc].u + 0, qr + kc * 32 + half * 8);
            load16B(qA[kc].u + 4, qr + kc * 32 + half * 8 + 16);
        }
    }

    v8f zero = {};
    v8f oacc[4];
    for (int f = 0; f < 4; ++f) oacc[f] = zero;
    float m_run[8], l_run[8];
    for (int r = 0; r < 8; ++r) { m_run[r] = -1e30f; l_run[r] = 0.0f; }

    for (int k0 = 0; k0 < S_LEN; k0 += 64) {
        __syncthreads();
        // stage K tile [k][d] via async direct-to-LDS; V tile transposed via VGPRs
        {
            int kr = tid >> 2;            // 0..63
            int dc = (tid & 3) << 4;      // 0,16,32,48
            const unsigned short* ksrc = Kh + (size_t)(k0 + kr) * DK + dc;
            unsigned short*       kdst = Kt + kr * 64 + dc;
            async_copy16B(kdst,     ksrc);
            async_copy16B(kdst + 8, ksrc + 8);

            const uint4* vs = reinterpret_cast<const uint4*>(Vh + (size_t)(k0 + kr) * DK + dc);
            uint4 va = vs[0], vb = vs[1];
            unsigned int vv[8] = { va.x, va.y, va.z, va.w, vb.x, vb.y, vb.z, vb.w };
            for (int j = 0; j < 8; ++j) {
                Vt[(dc + 2 * j + 0) * 64 + kr] = (unsigned short)(vv[j] & 0xFFFFu);
                Vt[(dc + 2 * j + 1) * 64 + kr] = (unsigned short)(vv[j] >> 16);
            }
        }
        wait_async0();
        __syncthreads();

        // logits = Q K^T + rel bias  (4 fragments of 16x16 per wave)
        v8f sfr[4];
        for (int f = 0; f < 4; ++f) {
            Frag16 kB[2];
            int n = f * 16 + l16;
            for (int kc = 0; kc < 2; ++kc) {
                load16B(kB[kc].u + 0, Kt + n * 64 + kc * 32 + half * 8);
                load16B(kB[kc].u + 4, Kt + n * 64 + kc * 32 + half * 8 + 16);
            }
            v8f s = zero;
            s = wmma_bf16(qA[0], kB[0], s);
            s = wmma_bf16(qA[1], kB[1], s);
            for (int r = 0; r < 8; ++r) {
                int ql = wave * 16 + r + half * 8;
                int gq = q0 + ql;
                int gk = k0 + f * 16 + l16;
                int dd = gk - gq;
                dd = dd < -16 ? -16 : (dd > 16 ? 16 : dd);
                s[r] += bf2f(relL[ql * NBUCKET + dd + 16]);
            }
            sfr[f] = s;
        }

        // online softmax update
        float sc[8];
        for (int r = 0; r < 8; ++r) {
            float mx = fmaxf(fmaxf(sfr[0][r], sfr[1][r]), fmaxf(sfr[2][r], sfr[3][r]));
            mx = redmax16(mx);
            float mnew = fmaxf(m_run[r], mx);
            sc[r] = __expf(m_run[r] - mnew);
            m_run[r] = mnew;
        }
        for (int f = 0; f < 4; ++f)
            for (int r = 0; r < 8; ++r) oacc[f][r] *= sc[r];
        if (l16 == 0) {  // one lane per q-row rescales its bucket sums
            for (int r = 0; r < 8; ++r) {
                int ql = wave * 16 + r + half * 8;
                for (int v = 0; v < NBUCKET; ++v) wsum[ql * NBUCKET + v] *= sc[r];
            }
        }

        float ladd[8];
        for (int r = 0; r < 8; ++r) ladd[r] = 0.0f;
        for (int f = 0; f < 4; ++f) {
            for (int r = 0; r < 8; ++r) {
                int ql = wave * 16 + r + half * 8;
                int gq = q0 + ql;
                int gk = k0 + f * 16 + l16;
                float p = __expf(sfr[f][r] - m_run[r]);
                ladd[r] += p;
                int dd = gk - gq;
                dd = dd < -16 ? -16 : (dd > 16 ? 16 : dd);
                atomicAdd(&wsum[ql * NBUCKET + dd + 16], p);
                Pst[wave][(r + half * 8) * 64 + f * 16 + l16] = f2bf(p);
            }
        }
        for (int r = 0; r < 8; ++r)
            l_run[r] = l_run[r] * sc[r] + redsum16(ladd[r]);

        __syncthreads();

        // O += P (16x64) * V (64x64)
        Frag16 pA[2];
        for (int kc = 0; kc < 2; ++kc) {
            const unsigned short* pr = &Pst[wave][l16 * 64 + kc * 32 + half * 8];
            load16B(pA[kc].u + 0, pr);
            load16B(pA[kc].u + 4, pr + 16);
        }
        for (int f = 0; f < 4; ++f) {
            int dcol = f * 16 + l16;
            Frag16 vB[2];
            for (int kc = 0; kc < 2; ++kc) {
                const unsigned short* vr = Vt + dcol * 64 + kc * 32 + half * 8;
                load16B(vB[kc].u + 0, vr);
                load16B(vB[kc].u + 4, vr + 16);
            }
            oacc[f] = wmma_bf16(pA[0], vB[0], oacc[f]);
            oacc[f] = wmma_bf16(pA[1], vB[1], oacc[f]);
        }
    }

    __syncthreads();

    // epilogue: ctx = (O + wsum @ embV) / l, store into (B,S,H*dk)
    const int b = bh >> 4;
    const int h = bh & 15;
    for (int f = 0; f < 4; ++f) {
        int d = f * 16 + l16;
        for (int r = 0; r < 8; ++r) {
            int ql = wave * 16 + r + half * 8;
            int gq = q0 + ql;
            float rv = 0.0f;
            for (int v = 0; v < NBUCKET; ++v)
                rv += wsum[ql * NBUCKET + v] * embV[v * DK + d];
            float val = (oacc[f][r] + rv) / l_run[r];
            ctx[((size_t)b * S_LEN + gq) * DMODEL + h * DK + d] = f2bf(val);
        }
    }
}

// ---------------- launcher ----------------

extern "C" void kernel_launch(void* const* d_in, const int* in_sizes, int n_in,
                              void* d_out, int out_size, void* d_ws, size_t ws_size,
                              hipStream_t stream) {
    const float* query = (const float*)d_in[0];
    const float* key_  = (const float*)d_in[1];
    const float* value = (const float*)d_in[2];
    const float* Wq    = (const float*)d_in[3];
    const float* Wk    = (const float*)d_in[4];
    const float* Wv    = (const float*)d_in[5];
    const float* Wo    = (const float*)d_in[6];
    const float* bq    = (const float*)d_in[7];
    const float* bk    = (const float*)d_in[8];
    const float* bv    = (const float*)d_in[9];
    const float* bo    = (const float*)d_in[10];
    const float* embK  = (const float*)d_in[11];
    const float* embV  = (const float*)d_in[12];

    const size_t NACT = (size_t)NB * S_LEN * DMODEL;  // 4096*1024
    const size_t NWGT = (size_t)DMODEL * DMODEL;      // 1024*1024

    unsigned short* qx  = (unsigned short*)d_ws;
    unsigned short* kx  = qx  + NACT;
    unsigned short* vx  = kx  + NACT;
    unsigned short* wqb = vx  + NACT;
    unsigned short* wkb = wqb + NWGT;
    unsigned short* wvb = wkb + NWGT;
    unsigned short* wob = wvb + NWGT;
    unsigned short* Qp  = wob + NWGT;
    unsigned short* Kp  = Qp  + NACT;
    unsigned short* Vp  = Kp  + NACT;
    unsigned short* ctx = Vp  + NACT;

    const int na = (int)NACT, nw = (int)NWGT;
    cvt_bf16_kernel<<<(na + 255) / 256, 256, 0, stream>>>(query, qx, na);
    cvt_bf16_kernel<<<(na + 255) / 256, 256, 0, stream>>>(key_,  kx, na);
    cvt_bf16_kernel<<<(na + 255) / 256, 256, 0, stream>>>(value, vx, na);
    cvt_bf16_kernel<<<(nw + 255) / 256, 256, 0, stream>>>(Wq, wqb, nw);
    cvt_bf16_kernel<<<(nw + 255) / 256, 256, 0, stream>>>(Wk, wkb, nw);
    cvt_bf16_kernel<<<(nw + 255) / 256, 256, 0, stream>>>(Wv, wvb, nw);
    cvt_bf16_kernel<<<(nw + 255) / 256, 256, 0, stream>>>(Wo, wob, nw);

    dim3 gg(DMODEL / 128, (NB * S_LEN) / 128);   // (8, 32)
    gemm_bf16_kernel<<<gg, 256, 0, stream>>>(qx, wqb, bq, Qp, nullptr,
                                             NB * S_LEN, DMODEL, DMODEL, 0);
    gemm_bf16_kernel<<<gg, 256, 0, stream>>>(kx, wkb, bk, Kp, nullptr,
                                             NB * S_LEN, DMODEL, DMODEL, 0);
    gemm_bf16_kernel<<<gg, 256, 0, stream>>>(vx, wvb, bv, Vp, nullptr,
                                             NB * S_LEN, DMODEL, DMODEL, 0);

    dim3 ga(S_LEN / 128, NB * NH);               // (16, 32)
    attn_rpr_kernel<<<ga, 256, 0, stream>>>(Qp, Kp, Vp, embK, embV, ctx);

    gemm_bf16_kernel<<<gg, 256, 0, stream>>>(ctx, wob, bo, nullptr, (float*)d_out,
                                             NB * S_LEN, DMODEL, DMODEL, 1);
}
